// ISTB_45689862095564
// MI455X (gfx1250) — compile-verified, matrix operand order checked by reading, not run on hardware
//
#include <hip/hip_runtime.h>

typedef float v8f __attribute__((ext_vector_type(8)));
typedef _Float16 v16h __attribute__((ext_vector_type(16)));

#define DIM 256
#define HEADS 8
#define HD 32
#define NTOK 585
#define NKT 37             // 16-key tiles (592)
#define NCH 19             // 32-key chunks (608)
#define NPAD2 608
#define TOTTOK (128*585)   // 74880
#define MBLK (TOTTOK/16)   // 4680

__device__ __forceinline__ v8f wmma16(v16h a, v16h b, v8f c) {
  return __builtin_amdgcn_wmma_f32_16x16x32_f16(false, a, false, b, (short)0, c, false, false);
}

__device__ __forceinline__ unsigned int pack2(float a, float b) {
  union { unsigned int u; _Float16 h[2]; } p;
  p.h[0] = (_Float16)a; p.h[1] = (_Float16)b;
  return p.u;
}

// K-pair selector for the 16-bit 16x16x32 operand layout:
// lanes 0-15 hold K-pairs {0..3, 8..11}, lanes 16-31 hold {4..7, 12..15}
__device__ __forceinline__ int ksel(int v, int hi4) { return hi4 + (v < 4 ? v : v + 4); }

// build a v16h fragment from 8 K-pair uints held in an LDS row (unit pair stride)
__device__ __forceinline__ v16h frag_row(const unsigned int* row, int pb, int hi4) {
  union { v16h v; unsigned int u[8]; } r;
#pragma unroll
  for (int v = 0; v < 8; ++v) r.u[v] = row[pb + ksel(v, hi4)];
  return r.v;
}

// ---------------- pack f32 weight [K][N] into f16 K-pair-major [K/2][N] uints ----------------
__global__ void packw_kernel(const float* __restrict__ W, unsigned int* __restrict__ Wp,
                             int ncols, int npairs) {
  int idx = blockIdx.x * blockDim.x + threadIdx.x;
  if (idx >= npairs * ncols) return;
  int p = idx / ncols, n = idx % ncols;
  Wp[idx] = pack2(W[(2 * p) * ncols + n], W[(2 * p + 1) * ncols + n]);
}

// ---------------- embed: per-scale channel->DIM projection + window partition ----------------
__global__ void embed_kernel(const float* __restrict__ x, const float* __restrict__ fcw,
                             const float* __restrict__ fcb, float* __restrict__ X,
                             int c, int win, int sbase) {
  int sp = 4 * win, sp3 = sp * sp * sp;
  int s = blockIdx.x;
  int ww = s % win; s /= win;
  int wh = s % win; s /= win;
  int wd = s % win; s /= win;
  int nw = s & 3; s >>= 2;
  int nh = s & 3; s >>= 2;
  int nd = s & 3; s >>= 2;
  int b = s;
  int d = nd * win + wd, h = nh * win + wh, w = nw * win + ww;
  __shared__ float lx[256];
  int tid = threadIdx.x;
  if (tid < c) lx[tid] = x[(size_t)(b * c + tid) * sp3 + (size_t)(d * sp + h) * sp + w];
  __syncthreads();
  float acc = fcb[tid];
  for (int ci = 0; ci < c; ++ci) acc += lx[ci] * fcw[ci * DIM + tid];
  int tok = sbase + (wd * win + wh) * win + ww;
  size_t row = (size_t)(((b * 4 + nd) * 4 + nh) * 4 + nw) * NTOK + tok;
  X[row * DIM + tid] = acc;
}

// ---------------- fused LayerNorm + QKV GEMM (f16 WMMA, f32 accum) ----------------
__global__ void qkv_kernel(const float* __restrict__ X, const unsigned int* __restrict__ Wp,
                           const float* __restrict__ bias, const float* __restrict__ lng,
                           const float* __restrict__ lnb, float* __restrict__ Out) {
  __shared__ float At[16][257];
  __shared__ unsigned int Ap[16][129];     // normalized tile, f16 K-pairs
  __shared__ float r1[16][16], r2[16][16];
  int tid = threadIdx.x;
  size_t m0 = (size_t)blockIdx.x * 16;
  for (int idx = tid; idx < 16 * DIM; idx += 256) {
    int m = idx >> 8, e = idx & 255;
    At[m][e] = X[(m0 + m) * DIM + e];
  }
  __syncthreads();
  {
    int m = tid >> 4, p = tid & 15;
    float s = 0.f, ss = 0.f;
    for (int e = p; e < DIM; e += 16) { float v = At[m][e]; s += v; ss += v * v; }
    r1[m][p] = s; r2[m][p] = ss;
  }
  __syncthreads();
  if ((tid & 15) == 0) {
    int m = tid >> 4;
    float s = 0.f, ss = 0.f;
    for (int p = 0; p < 16; ++p) { s += r1[m][p]; ss += r2[m][p]; }
    float mu = s * (1.f / DIM);
    float var = ss * (1.f / DIM) - mu * mu;
    r1[m][0] = mu;
    r2[m][0] = rsqrtf(var + 1e-6f);
  }
  __syncthreads();
  for (int idx = tid; idx < 16 * 128; idx += 256) {   // normalize + pack to f16 pairs
    int m = idx >> 7, p = idx & 127;
    float mu = r1[m][0], rs = r2[m][0];
    int e0 = 2 * p, e1 = 2 * p + 1;
    float v0 = (At[m][e0] - mu) * rs * lng[e0] + lnb[e0];
    float v1 = (At[m][e1] - mu) * rs * lng[e1] + lnb[e1];
    Ap[m][p] = pack2(v0, v1);
  }
  __syncthreads();
  int wave = tid >> 5, lane = tid & 31;
  int mrow = lane & 15, hi4 = (lane >> 4) << 2;
  for (int t = 0; t < 6; ++t) {                 // 48 N-tiles over 8 waves
    int n0 = (wave * 6 + t) * 16;
    int col = n0 + mrow;
    v8f acc = {};
    for (int kb = 0; kb < 8; ++kb) {            // 8 x (K=32) = 256
      int pb = kb * 16;
      v16h a = frag_row(&Ap[mrow][0], pb, hi4);
      union { v16h v; unsigned int u[8]; } bf;
#pragma unroll
      for (int v = 0; v < 8; ++v) bf.u[v] = Wp[(size_t)(pb + ksel(v, hi4)) * 768 + col];
      acc = wmma16(a, bf.v, acc);
    }
    float bv = bias[col];
    for (int r = 0; r < 8; ++r) {
      int m = ((lane >> 4) << 3) + r;
      Out[(m0 + m) * 768 + col] = acc[r] + bv;
    }
  }
}

// ---------------- attention: one (window, head, 16-query block) per workgroup ----------------
__global__ void attn_kernel(const float* __restrict__ QKV, float* __restrict__ AV) {
  __shared__ float S[16][NPAD2 + 1];           // f32 scores / probabilities
  __shared__ unsigned int Qtp[16][17];         // Q tile, f16 pairs over head-dim
  __shared__ unsigned int Kst[4][16][17];      // per-wave K tile (pairs over head-dim)
  __shared__ unsigned int Vst[4][16][33];      // per-wave V chunk (pairs over keys)
  __shared__ float Op[4][16][HD];
  __shared__ float r1[16][8], r2[16][8];
  int qb = blockIdx.x, head = blockIdx.y, win = blockIdx.z;
  int tid = threadIdx.x;                       // 128 threads = 4 waves
  int wave = tid >> 5, lane = tid & 31;
  size_t wbase = (size_t)win * NTOK;

  for (int idx = tid; idx < 16 * 16; idx += 128) {   // 16 rows x 16 d-pairs
    int m = idx >> 4, p = idx & 15;
    int gt = qb * 16 + m;
    float f0 = 0.f, f1 = 0.f;
    if (gt < NTOK) {
      size_t base = (wbase + gt) * 768 + head * HD + 2 * p;
      f0 = QKV[base]; f1 = QKV[base + 1];
    }
    Qtp[m][p] = pack2(f0, f1);
  }
  __syncthreads();

  int mrow = lane & 15, hi4 = (lane >> 4) << 2;
  v16h qf = frag_row(&Qtp[mrow][0], 0, hi4);   // Q fragment reused across all key tiles

  // phase 1: S = Q K^T  (one 16x16x32 WMMA per 16-key tile; K-dim = head dim)
  for (int kt = wave; kt < NKT; kt += 4) {
    for (int idx = lane; idx < 16 * 16; idx += 32) {
      int i = idx >> 4, p = idx & 15;
      int key = kt * 16 + i;
      float f0 = 0.f, f1 = 0.f;
      if (key < NTOK) {
        size_t base = (wbase + key) * 768 + 256 + head * HD + 2 * p;
        f0 = QKV[base]; f1 = QKV[base + 1];
      }
      Kst[wave][i][p] = pack2(f0, f1);
    }
    v16h bf = frag_row(&Kst[wave][mrow][0], 0, hi4);
    v8f acc = {};
    acc = wmma16(qf, bf, acc);
    for (int r = 0; r < 8; ++r) {
      int m = ((lane >> 4) << 3) + r;
      S[m][kt * 16 + mrow] = acc[r];
    }
  }
  __syncthreads();

  // phase 2: f32 row softmax (reference applies NO 1/sqrt(d) scale)
  {
    int m = tid & 15, p = tid >> 4;
    float mx = -3.0e38f;
    for (int n = p; n < NTOK; n += 8) mx = fmaxf(mx, S[m][n]);
    r1[m][p] = mx;
    __syncthreads();
    mx = r1[m][0];
    for (int q = 1; q < 8; ++q) mx = fmaxf(mx, r1[m][q]);
    float sum = 0.f;
    for (int n = p; n < NTOK; n += 8) {
      float e = __expf(S[m][n] - mx);
      S[m][n] = e;
      sum += e;
    }
    r2[m][p] = sum;
    __syncthreads();
    sum = 0.f;
    for (int q = 0; q < 8; ++q) sum += r2[m][q];
    float inv = 1.f / sum;
    for (int n = p; n < NTOK; n += 8) S[m][n] *= inv;
    if (p == 0) for (int n = NTOK; n < NPAD2; ++n) S[m][n] = 0.f;
  }
  __syncthreads();

  // phase 3: O = P @ V  (32 keys per WMMA, 19 chunks; per-wave partials)
  v8f acc0 = {}, acc1 = {};
  for (int ch = wave; ch < NCH; ch += 4) {
    for (int idx = lane; idx < 16 * HD; idx += 32) {   // 16 key-pairs x 32 dims
      int kp = idx >> 5, d = idx & 31;
      int k0 = ch * 32 + 2 * kp;
      float f0 = (k0 < NTOK)     ? QKV[(wbase + k0) * 768 + 512 + head * HD + d]     : 0.f;
      float f1 = (k0 + 1 < NTOK) ? QKV[(wbase + k0 + 1) * 768 + 512 + head * HD + d] : 0.f;
      Vst[wave][kp][d] = pack2(f0, f1);
    }
    union { v16h v; unsigned int u[8]; } af;
#pragma unroll
    for (int v = 0; v < 8; ++v) {
      int k0 = ch * 32 + 2 * ksel(v, hi4);
      af.u[v] = pack2(S[mrow][k0], S[mrow][k0 + 1]);
    }
    union { v16h v; unsigned int u[8]; } b0, b1;
#pragma unroll
    for (int v = 0; v < 8; ++v) {
      int t = ksel(v, hi4);
      b0.u[v] = Vst[wave][t][mrow];
      b1.u[v] = Vst[wave][t][16 + mrow];
    }
    acc0 = wmma16(af.v, b0.v, acc0);
    acc1 = wmma16(af.v, b1.v, acc1);
  }
  for (int r = 0; r < 8; ++r) {
    int m = ((lane >> 4) << 3) + r;
    Op[wave][m][mrow] = acc0[r];
    Op[wave][m][16 + mrow] = acc1[r];
  }
  __syncthreads();
  for (int idx = tid; idx < 16 * HD; idx += 128) {
    int m = idx >> 5, d = idx & 31;
    int gt = qb * 16 + m;
    if (gt < NTOK) {
      float s = Op[0][m][d] + Op[1][m][d] + Op[2][m][d] + Op[3][m][d];
      AV[(wbase + gt) * DIM + head * HD + d] = s;
    }
  }
}

// ---------------- output projection + bias + residual (f16 WMMA), in-place over AV ----------------
__global__ void proj_kernel(const float* __restrict__ AVin, const unsigned int* __restrict__ Wp,
                            const float* __restrict__ bias, const float* __restrict__ Xpre,
                            float* __restrict__ Out) {
  __shared__ unsigned int Ap[16][129];
  int tid = threadIdx.x;
  size_t m0 = (size_t)blockIdx.x * 16;
  for (int idx = tid; idx < 16 * 128; idx += 256) {
    int m = idx >> 7, p = idx & 127;
    size_t base = (m0 + m) * DIM + 2 * p;
    Ap[m][p] = pack2(AVin[base], AVin[base + 1]);
  }
  __syncthreads();
  int wave = tid >> 5, lane = tid & 31;
  int mrow = lane & 15, hi4 = (lane >> 4) << 2;
  for (int t = 0; t < 2; ++t) {               // 16 N-tiles over 8 waves
    int n0 = (wave * 2 + t) * 16;
    int col = n0 + mrow;
    v8f acc = {};
    for (int kb = 0; kb < 8; ++kb) {
      int pb = kb * 16;
      v16h a = frag_row(&Ap[mrow][0], pb, hi4);
      union { v16h v; unsigned int u[8]; } bf;
#pragma unroll
      for (int v = 0; v < 8; ++v) bf.u[v] = Wp[(size_t)(pb + ksel(v, hi4)) * 256 + col];
      acc = wmma16(a, bf.v, acc);
    }
    float bv = bias[col];
    for (int r = 0; r < 8; ++r) {
      int m = ((lane >> 4) << 3) + r;
      size_t o = (m0 + m) * DIM + col;
      Out[o] = acc[r] + bv + Xpre[o];
    }
  }
}

// ---------------- window reverse + DIM->channel projection, to NCDHW ----------------
__global__ void rev_kernel(const float* __restrict__ Y, const float* __restrict__ revw,
                           const float* __restrict__ revb, float* __restrict__ T,
                           int c, int win, int sbase) {
  int sp = 4 * win, sp3 = sp * sp * sp;
  int s = blockIdx.x;
  int w = s % sp; s /= sp;
  int h = s % sp; s /= sp;
  int d = s % sp; s /= sp;
  int b = s;
  int nd = d / win, wd = d % win;
  int nh = h / win, wh = h % win;
  int nw = w / win, ww = w % win;
  int tok = sbase + (wd * win + wh) * win + ww;
  size_t row = (size_t)(((b * 4 + nd) * 4 + nh) * 4 + nw) * NTOK + tok;
  __shared__ float ly[256];
  int tid = threadIdx.x;
  ly[tid] = Y[row * DIM + tid];
  __syncthreads();
  if (tid < c) {
    float acc = revb[tid];
    for (int e = 0; e < DIM; ++e) acc += ly[e] * revw[e * c + tid];
    T[(size_t)(b * c + tid) * sp3 + (size_t)(d * sp + h) * sp + w] = acc;
  }
}

// ---------------- 3x3x3 SAME conv, no bias ----------------
__global__ void conv_kernel(const float* __restrict__ in, const float* __restrict__ wgt,
                            float* __restrict__ out, int c, int sp) {
  int sp3 = sp * sp * sp;
  long long total = 2LL * c * sp3;
  long long idx = (long long)blockIdx.x * blockDim.x + threadIdx.x;
  if (idx >= total) return;
  int w = (int)(idx % sp); long long t = idx / sp;
  int h = (int)(t % sp); t /= sp;
  int d = (int)(t % sp); t /= sp;
  int co = (int)(t % c); t /= c;
  int b = (int)t;
  float acc = 0.f;
  for (int ci = 0; ci < c; ++ci) {
    const float* ip = in + (size_t)(b * c + ci) * sp3;
    const float* wp = wgt + (size_t)(co * c + ci) * 27;
    for (int kd = 0; kd < 3; ++kd) {
      int dd = d + kd - 1; if (dd < 0 || dd >= sp) continue;
      for (int kh = 0; kh < 3; ++kh) {
        int hh = h + kh - 1; if (hh < 0 || hh >= sp) continue;
        for (int kw = 0; kw < 3; ++kw) {
          int ww = w + kw - 1; if (ww < 0 || ww >= sp) continue;
          acc += ip[(size_t)(dd * sp + hh) * sp + ww] * wp[kd * 9 + kh * 3 + kw];
        }
      }
    }
  }
  out[idx] = acc;
}

// ---------------- training-mode BN stats / apply ----------------
__global__ void bnstats_kernel(const float* __restrict__ x, float* __restrict__ stats,
                               int c, int sp3) {
  int ch = blockIdx.x;
  int tid = threadIdx.x;
  long long n = 2LL * sp3;
  float s = 0.f, ss = 0.f;
  for (long long i = tid; i < n; i += blockDim.x) {
    int b = (int)(i / sp3);
    int si = (int)(i % sp3);
    float v = x[(size_t)(b * c + ch) * sp3 + si];
    s += v; ss += v * v;
  }
  __shared__ float l1[256], l2[256];
  l1[tid] = s; l2[tid] = ss;
  __syncthreads();
  for (int off = 128; off > 0; off >>= 1) {
    if (tid < off) { l1[tid] += l1[tid + off]; l2[tid] += l2[tid + off]; }
    __syncthreads();
  }
  if (tid == 0) {
    float mean = l1[0] / (float)n;
    float var = l2[0] / (float)n - mean * mean;
    stats[ch * 2] = mean;
    stats[ch * 2 + 1] = var;
  }
}

__global__ void bnact_kernel(float* __restrict__ x, const float* __restrict__ stats,
                             const float* __restrict__ g, const float* __restrict__ bb,
                             int c, int sp3) {
  long long total = 2LL * c * sp3;
  long long idx = (long long)blockIdx.x * blockDim.x + threadIdx.x;
  if (idx >= total) return;
  int ch = (int)((idx / sp3) % c);
  float v = (x[idx] - stats[ch * 2]) * rsqrtf(stats[ch * 2 + 1] + 1e-5f) * g[ch] + bb[ch];
  x[idx] = (v >= 0.f) ? v : 0.01f * v;
}

__global__ void bnres_kernel(const float* __restrict__ x, const float* __restrict__ stats,
                             const float* __restrict__ g, const float* __restrict__ bb,
                             const float* __restrict__ t, float* __restrict__ out,
                             int c, int sp3) {
  long long total = 2LL * c * sp3;
  long long idx = (long long)blockIdx.x * blockDim.x + threadIdx.x;
  if (idx >= total) return;
  int ch = (int)((idx / sp3) % c);
  float v = (x[idx] - stats[ch * 2]) * rsqrtf(stats[ch * 2 + 1] + 1e-5f) * g[ch] + bb[ch] + t[idx];
  out[idx] = (v >= 0.f) ? v : 0.01f * v;
}

extern "C" void kernel_launch(void* const* d_in, const int* in_sizes, int n_in,
                              void* d_out, int out_size, void* d_ws, size_t ws_size,
                              hipStream_t stream) {
  (void)in_sizes; (void)n_in; (void)out_size; (void)ws_size;
  float* ws = (float*)d_ws;
  float* X   = ws;                      // 74880*256
  float* QKV = ws + 19169280LL;         // 74880*768
  float* AV  = ws + 76677120LL;         // 74880*256 (also holds Y after proj)
  float* T   = ws + 95846400LL;         // residual NCDHW, all scales (2785280)
  float* H1  = ws + 98631680LL;         // conv scratch (max scale size)
  float* H2  = ws + 100728832LL;
  float* ST  = ws + 102825984LL;        // BN stats
  unsigned int* WQp = (unsigned int*)(ws + 102826496LL);  // 128*768 packed f16 pairs
  unsigned int* WPp = (unsigned int*)(ws + 102924800LL);  // 128*256 packed f16 pairs

  const float* ln_g   = (const float*)d_in[44];
  const float* ln_b   = (const float*)d_in[45];
  const float* qkv_w  = (const float*)d_in[46];
  const float* qkv_b  = (const float*)d_in[47];
  const float* proj_w = (const float*)d_in[48];
  const float* proj_b = (const float*)d_in[49];

  const int wins[4]  = {8, 4, 2, 1};
  const int chs[4]   = {32, 64, 128, 256};
  const int sbase[4] = {0, 512, 576, 584};
  const long long toff[4] = {0, 2097152, 2621440, 2752512};

  packw_kernel<<<(128 * 768 + 255) / 256, 256, 0, stream>>>(qkv_w, WQp, 768, 128);
  packw_kernel<<<(128 * 256 + 255) / 256, 256, 0, stream>>>(proj_w, WPp, 256, 128);

  for (int j = 0; j < 4; ++j) {
    const float* xj  = (const float*)d_in[j * 11 + 0];
    const float* fcw = (const float*)d_in[j * 11 + 1];
    const float* fcb = (const float*)d_in[j * 11 + 2];
    int win = wins[j];
    int nblk = 2 * 64 * win * win * win;
    embed_kernel<<<nblk, 256, 0, stream>>>(xj, fcw, fcb, X, chs[j], win, sbase[j]);
  }

  qkv_kernel<<<MBLK, 256, 0, stream>>>(X, WQp, qkv_b, ln_g, ln_b, QKV);
  attn_kernel<<<dim3(37, 8, 128), 128, 0, stream>>>(QKV, AV);
  proj_kernel<<<MBLK, 256, 0, stream>>>(AV, WPp, proj_b, X, AV);

  for (int j = 0; j < 4; ++j) {
    const float* revw = (const float*)d_in[j * 11 + 3];
    const float* revb = (const float*)d_in[j * 11 + 4];
    const float* cw1  = (const float*)d_in[j * 11 + 5];
    const float* cw2  = (const float*)d_in[j * 11 + 6];
    const float* g1   = (const float*)d_in[j * 11 + 7];
    const float* b1   = (const float*)d_in[j * 11 + 8];
    const float* g2   = (const float*)d_in[j * 11 + 9];
    const float* b2   = (const float*)d_in[j * 11 + 10];
    int win = wins[j], c = chs[j], sp = 4 * win, sp3 = sp * sp * sp;
    float* Tj = T + toff[j];
    float* outj = (float*)d_out + toff[j];
    long long total = 2LL * c * sp3;
    int grid = (int)((total + 255) / 256);

    rev_kernel<<<2 * sp3, 256, 0, stream>>>(AV, revw, revb, Tj, c, win, sbase[j]);
    conv_kernel<<<grid, 256, 0, stream>>>(Tj, cw1, H1, c, sp);
    bnstats_kernel<<<c, 256, 0, stream>>>(H1, ST, c, sp3);
    bnact_kernel<<<grid, 256, 0, stream>>>(H1, ST, g1, b1, c, sp3);
    conv_kernel<<<grid, 256, 0, stream>>>(H1, cw2, H2, c, sp);
    bnstats_kernel<<<c, 256, 0, stream>>>(H2, ST, c, sp3);
    bnres_kernel<<<grid, 256, 0, stream>>>(H2, ST, g2, b2, Tj, outj, c, sp3);
  }
}
